// MeshGraphSAGE_11106785428114
// MI455X (gfx1250) — matrix operand order repaired and checked by reading, not compiled, not run on hardware
//
#include <hip/hip_runtime.h>

#define NN 50000
#define NE 800000
#define E2T (NE + NN)      // 850000 edges incl. self loops
#define HID 128
#define HEADS 4

typedef __attribute__((ext_vector_type(16))) __bf16        v16bf;
typedef __attribute__((ext_vector_type(8)))  float         v8f;
typedef __attribute__((ext_vector_type(4)))  unsigned int  u32x4;
typedef __attribute__((ext_vector_type(4)))  float         f32x4;

union BF16Tile { u32x4 q[2]; v16bf v; };

__device__ __forceinline__ unsigned short f2bf(float f) {
    union { float f; unsigned u; } v; v.f = f;
    unsigned u = v.u;
    u += 0x7FFFu + ((u >> 16) & 1u);       // round-to-nearest-even
    return (unsigned short)(u >> 16);
}

// float atomic max via ordered-integer trick (init must be -large)
__device__ __forceinline__ void atomicMaxF(float* addr, float v) {
    if (v >= 0.0f) atomicMax((int*)addr, __float_as_int(v));
    else           atomicMin((unsigned int*)addr, __float_as_uint(v));
}

__global__ void fill_f32_kernel(float* __restrict__ p, long long n, float v) {
    long long i = (long long)blockIdx.x * blockDim.x + threadIdx.x;
    if (i < n) p[i] = v;
}

// ---- GAT input: h = x @ W_gat (K=4), plus per-(node,head) attention dots ----
__global__ __launch_bounds__(256)
void gat_input_kernel(const float* __restrict__ x, const float* __restrict__ Wg,
                      const float* __restrict__ atts, const float* __restrict__ attd,
                      float* __restrict__ h, float* __restrict__ a_s, float* __restrict__ a_d) {
    int n = blockIdx.x;
    int t = threadIdx.x;
    float x0 = x[n * 4 + 0], x1 = x[n * 4 + 1], x2 = x[n * 4 + 2], x3 = x[n * 4 + 3];
    __shared__ float rs[512];
    __shared__ float rd[512];
    #pragma unroll
    for (int j = 0; j < 2; ++j) {
        int c = t + j * 256;                       // 0..511
        float v = x0 * Wg[c] + x1 * Wg[512 + c] + x2 * Wg[1024 + c] + x3 * Wg[1536 + c];
        h[(long long)n * 512 + c] = v;
        int head = c >> 7, cc = c & 127;
        rs[c] = v * atts[head * 128 + cc];
        rd[c] = v * attd[head * 128 + cc];
    }
    __syncthreads();
    int seg = t >> 6, li = t & 63;                 // 4 segments of 128, 64 reducers each
    for (int st = 64; st > 0; st >>= 1) {
        if (li < st) {
            rs[seg * 128 + li] += rs[seg * 128 + li + st];
            rd[seg * 128 + li] += rd[seg * 128 + li + st];
        }
        __syncthreads();
    }
    if (li == 0) {
        a_s[n * 4 + seg] = rs[seg * 128];
        a_d[n * 4 + seg] = rd[seg * 128];
    }
}

// ---- edge attention logits + segment max ----
__global__ void edge_logits_kernel(const long long* __restrict__ srcp, const long long* __restrict__ dstp,
                                   const float* __restrict__ a_s, const float* __restrict__ a_d,
                                   float* __restrict__ eexp, float* __restrict__ mbuf) {
    int i = blockIdx.x * blockDim.x + threadIdx.x;
    if (i >= E2T) return;
    long long s, d;
    if (i < NE) { s = srcp[i]; d = dstp[i]; } else { s = d = i - NE; }
    #pragma unroll
    for (int hd = 0; hd < HEADS; ++hd) {
        float ev = a_s[s * 4 + hd] + a_d[d * 4 + hd];
        ev = ev > 0.0f ? ev : 0.2f * ev;          // leaky relu
        eexp[(long long)i * 4 + hd] = ev;         // store logit for now
        atomicMaxF(&mbuf[d * 4 + hd], ev);
    }
}

// ---- exp(e - max) + segment sum ----
__global__ void edge_exp_kernel(const long long* __restrict__ srcp, const long long* __restrict__ dstp,
                                const float* __restrict__ mbuf,
                                float* __restrict__ eexp, float* __restrict__ zbuf) {
    int i = blockIdx.x * blockDim.x + threadIdx.x;
    if (i >= E2T) return;
    long long d;
    if (i < NE) { d = dstp[i]; } else { d = i - NE; }
    #pragma unroll
    for (int hd = 0; hd < HEADS; ++hd) {
        float v = __expf(eexp[(long long)i * 4 + hd] - mbuf[d * 4 + hd]);
        eexp[(long long)i * 4 + hd] = v;
        atomicAdd(&zbuf[d * 4 + hd], v);
    }
}

// ---- weighted message scatter, head-mean folded in (factor 0.25) ----
__global__ __launch_bounds__(128)
void gat_message_kernel(const long long* __restrict__ srcp, const long long* __restrict__ dstp,
                        const float* __restrict__ h, const float* __restrict__ eexp,
                        const float* __restrict__ zbuf, float* __restrict__ outm) {
    int e = blockIdx.x;                            // 0..E2T-1
    int c = threadIdx.x;                           // 0..127
    long long s, d;
    if (e < NE) { s = srcp[e]; d = dstp[e]; } else { s = d = e - NE; }
    float acc = 0.0f;
    #pragma unroll
    for (int hd = 0; hd < HEADS; ++hd) {
        float alpha = eexp[(long long)e * 4 + hd] / zbuf[d * 4 + hd];
        acc += alpha * h[(long long)s * 512 + hd * 128 + c];
    }
    atomicAdd(&outm[(long long)d * 128 + c], 0.25f * acc);
}

// ---- GAT finalize: relu(mean + bias), emit f32 + bf16 copies ----
__global__ void gat_finalize_kernel(const float* __restrict__ outm, const float* __restrict__ bgat,
                                    float* __restrict__ h1f, unsigned short* __restrict__ h1bf) {
    long long i = (long long)blockIdx.x * blockDim.x + threadIdx.x;
    if (i >= (long long)NN * 128) return;
    int c = (int)(i & 127);
    float v = outm[i] + bgat[c];
    v = v > 0.0f ? v : 0.0f;
    h1f[i] = v;
    h1bf[i] = f2bf(v);
}

// ---- SAGE neighbor sum + degree ----
__global__ __launch_bounds__(128)
void sage_agg_kernel(const long long* __restrict__ srcp, const long long* __restrict__ dstp,
                     const float* __restrict__ h1f, float* __restrict__ S, float* __restrict__ deg) {
    int e = blockIdx.x;                            // 0..NE-1
    int c = threadIdx.x;
    long long s = srcp[e], d = dstp[e];
    atomicAdd(&S[(long long)d * 128 + c], h1f[(long long)s * 128 + c]);
    if (c == 0) atomicAdd(&deg[d], 1.0f);
}

// ---- SAGE mean -> bf16 ----
__global__ void sage_mean_kernel(const float* __restrict__ S, const float* __restrict__ deg,
                                 unsigned short* __restrict__ meanbf) {
    long long i = (long long)blockIdx.x * blockDim.x + threadIdx.x;
    if (i >= (long long)NN * 128) return;
    float dg = deg[i >> 7];
    float v = dg > 0.0f ? S[i] / dg : 0.0f;
    meanbf[i] = f2bf(v);
}

// ---- pack a 128x128 f32 weight (row-major, K x N) into WMMA B-operand bf16 layout ----
// Per (kt,nt) 32x16 tile: lane = half*16 + (n & 15); element j (0..15): K = kt*32 + (j < 8 ? half*8 + j : 16 + half*8 + (j - 8))
__global__ void pack_weights_kernel(const float* __restrict__ W, unsigned short* __restrict__ P) {
    int idx = blockIdx.x * blockDim.x + threadIdx.x;   // 0..16383
    if (idx >= 128 * 128) return;
    int j    = idx & 15;
    int lane = (idx >> 4) & 31;
    int tile = idx >> 9;                               // kt*8 + nt
    int kt = tile >> 3, nt = tile & 7;
    int n  = nt * 16 + (lane & 15);
    int half = lane >> 4;
    int K = kt * 32 + (j < 8 ? half * 8 + j : 16 + half * 8 + (j - 8));
    P[idx] = f2bf(W[K * 128 + n]);
}

// ---- fused SAGE GEMMs: h2 = relu(mean@Wl + h1@Wr + b), bf16 WMMA, f32 accumulate ----
__global__ __launch_bounds__(256)
void sage_gemm_kernel(const unsigned short* __restrict__ meanbf, const unsigned short* __restrict__ h1bf,
                      const unsigned short* __restrict__ packBl, const unsigned short* __restrict__ packBr,
                      const float* __restrict__ bsage, float* __restrict__ h2) {
    int wid  = (blockIdx.x * blockDim.x + threadIdx.x) >> 5;
    int lane = threadIdx.x & 31;
    long long r0 = (long long)wid * 16;
    if (r0 >= NN) return;                               // wave-uniform
    int m = lane & 15, half = lane >> 4;

    v8f acc[8];
    #pragma unroll
    for (int nt = 0; nt < 8; ++nt) {
        float b = bsage[nt * 16 + m];
        #pragma unroll
        for (int t = 0; t < 8; ++t) acc[nt][t] = b;
    }

    #pragma unroll
    for (int kt = 0; kt < 4; ++kt) {
        // A tiles (16x32 bf16): lane holds row r0+m; regs 0-3 = K half*8+0..7, regs 4-7 = K 16+half*8+0..7
        BF16Tile a1, a2;
        const unsigned short* pa = meanbf + (r0 + m) * 128 + kt * 32 + half * 8;
        a1.q[0] = *(const u32x4*)(pa);
        a1.q[1] = *(const u32x4*)(pa + 16);
        const unsigned short* pb = h1bf + (r0 + m) * 128 + kt * 32 + half * 8;
        a2.q[0] = *(const u32x4*)(pb);
        a2.q[1] = *(const u32x4*)(pb + 16);
        #pragma unroll
        for (int nt = 0; nt < 8; ++nt) {
            BF16Tile w1, w2;
            const unsigned short* pw1 = packBl + ((kt * 8 + nt) * 32 + lane) * 16;
            w1.q[0] = ((const u32x4*)pw1)[0];
            w1.q[1] = ((const u32x4*)pw1)[1];
            const unsigned short* pw2 = packBr + ((kt * 8 + nt) * 32 + lane) * 16;
            w2.q[0] = ((const u32x4*)pw2)[0];
            w2.q[1] = ((const u32x4*)pw2)[1];
            acc[nt] = __builtin_amdgcn_wmma_f32_16x16x32_bf16(false, a1.v, false, w1.v,
                                                              (short)0, acc[nt], false, false);
            acc[nt] = __builtin_amdgcn_wmma_f32_16x16x32_bf16(false, a2.v, false, w2.v,
                                                              (short)0, acc[nt], false, false);
        }
    }
    // C/D layout: reg v -> row r0 + v + half*8, col nt*16 + m
    #pragma unroll
    for (int nt = 0; nt < 8; ++nt) {
        #pragma unroll
        for (int v = 0; v < 8; ++v) {
            long long row = r0 + v + half * 8;
            int col = nt * 16 + m;
            float xv = acc[nt][v];
            h2[row * 128 + col] = xv > 0.0f ? xv : 0.0f;
        }
    }
}

// ---- FC + sigmoid: one wave per row, lane handles 4 channels, shuffle-reduce ----
__global__ __launch_bounds__(256)
void fc_kernel(const float* __restrict__ h2, const float* __restrict__ Wfc,
               const float* __restrict__ bfc, float* __restrict__ out) {
    int wid  = (blockIdx.x * blockDim.x + threadIdx.x) >> 5;
    int lane = threadIdx.x & 31;
    if (wid >= NN) return;
    f32x4 a = *(const f32x4*)(h2 + (long long)wid * 128 + lane * 4);
    f32x4 w = *(const f32x4*)(Wfc + lane * 4);
    float dot = a.x * w.x + a.y * w.y + a.z * w.z + a.w * w.w;
    #pragma unroll
    for (int off = 16; off > 0; off >>= 1) dot += __shfl_xor(dot, off, 32);
    if (lane == 0) out[wid] = 1.0f / (1.0f + __expf(-(dot + bfc[0])));
}

extern "C" void kernel_launch(void* const* d_in, const int* in_sizes, int n_in,
                              void* d_out, int out_size, void* d_ws, size_t ws_size,
                              hipStream_t stream) {
    const float*     x     = (const float*)d_in[0];
    const long long* eidx  = (const long long*)d_in[1];
    const float*     Wgat  = (const float*)d_in[2];
    const float*     atts  = (const float*)d_in[3];
    const float*     attd  = (const float*)d_in[4];
    const float*     bgat  = (const float*)d_in[5];
    const float*     Wsl   = (const float*)d_in[6];
    const float*     bsage = (const float*)d_in[7];
    const float*     Wsr   = (const float*)d_in[8];
    const float*     Wfc   = (const float*)d_in[9];
    const float*     bfc   = (const float*)d_in[10];
    float*           out   = (float*)d_out;

    const long long* srcp = eidx;
    const long long* dstp = eidx + NE;

    size_t off = 0;
    auto alloc = [&](size_t bytes) -> char* {
        char* p = (char*)d_ws + off;
        off += (bytes + 255) & ~(size_t)255;
        return p;
    };
    float*          h      = (float*)alloc((size_t)NN * 512 * 4);   // region A (reused below)
    float*          outm   = (float*)alloc((size_t)NN * 128 * 4);
    float*          a_s    = (float*)alloc((size_t)NN * 4 * 4);
    float*          a_d    = (float*)alloc((size_t)NN * 4 * 4);
    float*          mbuf   = (float*)alloc((size_t)NN * 4 * 4);
    float*          zbuf   = (float*)alloc((size_t)NN * 4 * 4);
    float*          eexp   = (float*)alloc((size_t)E2T * 4 * 4);
    float*          h1f    = (float*)alloc((size_t)NN * 128 * 4);
    unsigned short* h1bf   = (unsigned short*)alloc((size_t)NN * 128 * 2);
    unsigned short* packBl = (unsigned short*)alloc((size_t)128 * 128 * 2);
    unsigned short* packBr = (unsigned short*)alloc((size_t)128 * 128 * 2);
    // reuse region A (h, 102MB) once the message phase no longer reads it:
    float*          S      = h;                                     // 25.6MB
    float*          deg    = h + (size_t)NN * 128;                  // 0.2MB
    unsigned short* meanbf = (unsigned short*)(h + (size_t)NN * 128 + NN);
    float*          h2     = (float*)((char*)meanbf + (size_t)NN * 128 * 2);

    const int T = 256;
    // init
    fill_f32_kernel<<<(NN * 4 + T - 1) / T, T, 0, stream>>>(mbuf, (long long)NN * 4, -1e30f);
    fill_f32_kernel<<<(NN * 4 + T - 1) / T, T, 0, stream>>>(zbuf, (long long)NN * 4, 0.0f);
    fill_f32_kernel<<<((long long)NN * 128 + T - 1) / T, T, 0, stream>>>(outm, (long long)NN * 128, 0.0f);

    // GAT
    gat_input_kernel<<<NN, 256, 0, stream>>>(x, Wgat, atts, attd, h, a_s, a_d);
    edge_logits_kernel<<<(E2T + T - 1) / T, T, 0, stream>>>(srcp, dstp, a_s, a_d, eexp, mbuf);
    edge_exp_kernel<<<(E2T + T - 1) / T, T, 0, stream>>>(srcp, dstp, mbuf, eexp, zbuf);
    gat_message_kernel<<<E2T, 128, 0, stream>>>(srcp, dstp, h, eexp, zbuf, outm);
    gat_finalize_kernel<<<((long long)NN * 128 + T - 1) / T, T, 0, stream>>>(outm, bgat, h1f, h1bf);

    // SAGE aggregate (region A now reusable)
    fill_f32_kernel<<<((long long)NN * 128 + T - 1) / T, T, 0, stream>>>(S, (long long)NN * 128, 0.0f);
    fill_f32_kernel<<<(NN + T - 1) / T, T, 0, stream>>>(deg, (long long)NN, 0.0f);
    sage_agg_kernel<<<NE, 128, 0, stream>>>(srcp, dstp, h1f, S, deg);
    sage_mean_kernel<<<((long long)NN * 128 + T - 1) / T, T, 0, stream>>>(S, deg, meanbf);

    // weights -> WMMA B layout
    pack_weights_kernel<<<64, 256, 0, stream>>>(Wsl, packBl);
    pack_weights_kernel<<<64, 256, 0, stream>>>(Wsr, packBr);

    // fused SAGE GEMMs (bf16 WMMA) + relu
    int nwaves = NN / 16;                                  // 3125
    sage_gemm_kernel<<<(nwaves + 7) / 8, 256, 0, stream>>>(meanbf, h1bf, packBl, packBr, bsage, h2);

    // FC + sigmoid
    fc_kernel<<<(NN + 7) / 8, 256, 0, stream>>>(h2, Wfc, bfc, out);
}
